// FluxAttention_85521388798627
// MI455X (gfx1250) — matrix operand order repaired and checked by reading, not compiled, not run on hardware
//
#include <hip/hip_runtime.h>
#include <hip/hip_bf16.h>
#include <stdint.h>

#define SQ   4096
#define DIMC 3072
#define NH   24
#define HD   128
#define SCALE 0.08838834764831845f   // 1/sqrt(128)
#define PAD  8                       // 16B row pad -> 80B/272B pitches, bank-friendly

typedef __attribute__((ext_vector_type(16))) __bf16 v16bf;
typedef __attribute__((ext_vector_type(8)))  float  v8f;

union Frag { v16bf v; uint4 q[2]; uint32_t u[8]; };

// ---- Tensor Data Mover support (probe-verified builtin, guarded) ----------
#if defined(__has_builtin)
#if __has_builtin(__builtin_amdgcn_tensor_load_to_lds) && \
    __has_builtin(__builtin_amdgcn_s_wait_tensorcnt)
#define USE_TDM 1
#endif
#endif
#ifndef USE_TDM
#define USE_TDM 0
#endif

#if USE_TDM
typedef __attribute__((ext_vector_type(4))) unsigned int tdm_v4u;
typedef __attribute__((ext_vector_type(8))) int          tdm_v8i;
typedef __attribute__((ext_vector_type(4))) int          tdm_v4i;

// 2D tile load global->LDS via TDM (D# per CDNA5 ISA).
// tile0 = elements per row, tile1 = rows, stride0 = tensor dim0 stride (elements),
// pad codes: interval 2^(c+1) DWORDs between pads, amount (c+1) DWORDs inserted.
__device__ __forceinline__ void tdm_load_2d(uint32_t lds_off, const void* gptr,
                                            uint32_t tile0, uint32_t tile1,
                                            uint64_t stride0,
                                            uint32_t pad_int, uint32_t pad_amt) {
    const uint64_t ga = (uint64_t)(uintptr_t)gptr;
    tdm_v4u g0 = { 1u,                                   // count=1, user mode
                   lds_off,                              // LDS byte address
                   (uint32_t)ga,                         // global addr [31:0]
                   ((uint32_t)(ga >> 32) & 0x01FFFFFFu)  // global addr [56:32]
                     | (2u << 30) };                     // type = 2 ("image")
    const uint32_t d0 = tile0, d1 = tile1;               // tensor dims == tile dims
    tdm_v8i g1 = { (int)((1u << 16)                      // data_size = 2 bytes
                         | (1u << 20)                    // pad_enable
                         | (pad_int << 22) | (pad_amt << 25)),
                   (int)((d0 & 0xFFFFu) << 16),          // [47:32]=0, dim0 lo
                   (int)((d0 >> 16) | ((d1 & 0xFFFFu) << 16)),
                   (int)((d1 >> 16) | (tile0 << 16)),    // tile_dim0
                   (int)(tile1 & 0xFFFFu),               // tile_dim1 (tile_dim2=0)
                   (int)(uint32_t)(stride0 & 0xFFFFFFFFu),
                   (int)(uint32_t)((stride0 >> 32) & 0xFFFFu),
                   0 };
    tdm_v4i z4 = { 0, 0, 0, 0 };
#if __clang_major__ >= 23
    tdm_v8i z8 = { 0, 0, 0, 0, 0, 0, 0, 0 };
    __builtin_amdgcn_tensor_load_to_lds(g0, g1, z4, z4, z8, 0);
#else
    __builtin_amdgcn_tensor_load_to_lds(g0, g1, z4, z4, 0);
#endif
}
#endif  // USE_TDM

__device__ __forceinline__ float bf2f(uint16_t b) {
    union { uint32_t u; float f; } c; c.u = ((uint32_t)b) << 16; return c.f;
}
__device__ __forceinline__ uint16_t f2bf(float f) {
    union { float f; uint32_t u; } c; c.f = f;
    uint32_t r = c.u + 0x7FFFu + ((c.u >> 16) & 1u);
    return (uint16_t)(r >> 16);
}
__device__ __forceinline__ v8f vzero8() {
    v8f z;
#pragma unroll
    for (int i = 0; i < 8; ++i) z[i] = 0.0f;
    return z;
}
__device__ __forceinline__ v8f wmma_bf16(v16bf a, v16bf b, v8f c) {
    return __builtin_amdgcn_wmma_f32_16x16x32_bf16(false, a, false, b, (short)0, c, false, false);
}

// ---------------------------------------------------------------- fp32 -> bf16
__global__ void f32_to_bf16_kernel(const float* __restrict__ in,
                                   uint16_t* __restrict__ out, int n) {
    int i = blockIdx.x * blockDim.x + threadIdx.x;
    if (i < n) out[i] = f2bf(in[i]);
}

// -------------------------------------------- fp32 [R][C] -> bf16 [C][R] (W^T)
__global__ __launch_bounds__(256) void transpose_f32_to_bf16(
    const float* __restrict__ in, uint16_t* __restrict__ out, int R, int C) {
    __shared__ float tile[32][33];
    const int t  = threadIdx.x;
    const int tx = t & 31, ty = t >> 5;          // ty: 0..7
    const int r0 = blockIdx.y * 32;
    const int c0 = blockIdx.x * 32;
#pragma unroll
    for (int i = 0; i < 4; ++i)
        tile[ty + i * 8][tx] = in[(size_t)(r0 + ty + i * 8) * C + c0 + tx];
    __syncthreads();
#pragma unroll
    for (int i = 0; i < 4; ++i)
        out[(size_t)(c0 + ty + i * 8) * R + r0 + tx] = f2bf(tile[tx][ty + i * 8]);
}

// -------------------------------------------- bf16 [R][C] -> bf16 [C][R] (V^T)
__global__ __launch_bounds__(256) void transpose_bf16(
    const uint16_t* __restrict__ in, uint16_t* __restrict__ out, int R, int C) {
    __shared__ uint16_t tile[32][34];
    const int t  = threadIdx.x;
    const int tx = t & 31, ty = t >> 5;
    const int r0 = blockIdx.y * 32;
    const int c0 = blockIdx.x * 32;
#pragma unroll
    for (int i = 0; i < 4; ++i)
        tile[ty + i * 8][tx] = in[(size_t)(r0 + ty + i * 8) * C + c0 + tx];
    __syncthreads();
#pragma unroll
    for (int i = 0; i < 4; ++i)
        out[(size_t)(c0 + ty + i * 8) * R + r0 + tx] = tile[tx][ty + i * 8];
}

// ---------------------------------------------------------------- GEMM + bias
// C[m,n] = sum_k X[m,k] * Wt[n,k] + bias[n]. Block tile 64(M) x 128(N), Kstep 32.
// 256 thr / 8 waves; each wave: 1 M-subtile x 4 N-subtiles. Every WMMA operand
// fragment is two contiguous ds_load_b128 from the K-innermost LDS tiles.
__global__ __launch_bounds__(256) void gemm_bias_bf16(
    const uint16_t* __restrict__ X, const uint16_t* __restrict__ Wt,
    const float* __restrict__ bias, uint16_t* __restrict__ out) {
    __shared__ uint16_t Xs [64][32 + PAD];
    __shared__ uint16_t Wst[128][32 + PAD];
    const int t    = threadIdx.x;
    const int lane = t & 31;
    const int wid  = t >> 5;
    const int ln   = lane & 15;
    const int hi   = lane >> 4;
    const int m0   = blockIdx.x * 64;
    const int n0   = blockIdx.y * 128;
    const int mt   = wid >> 1;            // 0..3
    const int nt0  = (wid & 1) * 4;       // 0 or 4

    v8f acc[4];
#pragma unroll
    for (int i = 0; i < 4; ++i) acc[i] = vzero8();

    const int xr = t >> 2;                // 0..63
    const int xc = (t & 3) * 8;           // 0..24
    const int wr = t >> 1;                // 0..127
    const int wc = (t & 1) * 16;          // 0 or 16

    for (int k0 = 0; k0 < DIMC; k0 += 32) {
        *(uint4*)&Xs[xr][xc] = *(const uint4*)&X[(size_t)(m0 + xr) * DIMC + k0 + xc];
        const size_t wb = (size_t)(n0 + wr) * DIMC + k0 + wc;
        *(uint4*)&Wst[wr][wc]     = *(const uint4*)&Wt[wb];
        *(uint4*)&Wst[wr][wc + 8] = *(const uint4*)&Wt[wb + 8];
        __syncthreads();

        Frag A;
        const int am = mt * 16 + ln;
        const int kh = hi * 16;
        A.q[0] = *(const uint4*)&Xs[am][kh];
        A.q[1] = *(const uint4*)&Xs[am][kh + 8];
        const int khB = hi * 16;
#pragma unroll
        for (int nt = 0; nt < 4; ++nt) {
            Frag B;
            const int n = (nt0 + nt) * 16 + ln;
            B.q[0] = *(const uint4*)&Wst[n][khB];
            B.q[1] = *(const uint4*)&Wst[n][khB + 8];
            acc[nt] = wmma_bf16(A.v, B.v, acc[nt]);
        }
        __syncthreads();
    }
#pragma unroll
    for (int nt = 0; nt < 4; ++nt)
#pragma unroll
        for (int j = 0; j < 8; ++j) {
            const int m = m0 + mt * 16 + j + hi * 8;
            const int n = n0 + (nt0 + nt) * 16 + ln;
            out[(size_t)m * DIMC + n] = f2bf(acc[nt][j] + bias[n]);
        }
}

// ------------------------------------------------- RMSNorm + RoPE (in place)
__global__ __launch_bounds__(128) void rmsnorm_rope_kernel(
    uint16_t* __restrict__ X, const float* __restrict__ g,
    const float* __restrict__ cosb, const float* __restrict__ sinb) {
    __shared__ float red[128];
    __shared__ float rowv[128];
    const int rid = blockIdx.x;          // s*NH + h
    const int s   = rid / NH;
    const int d   = threadIdx.x;
    const size_t base = (size_t)rid * HD;
    float x = bf2f(X[base + d]);
    red[d] = x * x;
    __syncthreads();
#pragma unroll
    for (int off = 64; off > 0; off >>= 1) {
        if (d < off) red[d] += red[d + off];
        __syncthreads();
    }
    const float r  = rsqrtf(red[0] * (1.0f / 128.0f) + 1e-6f);
    const float xn = x * r * g[d];
    rowv[d] = xn;
    __syncthreads();
    const float xr = (d & 1) ? rowv[d - 1] : -rowv[d + 1];
    const size_t cs = (size_t)s * HD + d;
    X[base + d] = f2bf(xn * cosb[cs] + xr * sinb[cs]);
}

// ----------------------------------------------------------- flash attention
// Block: 256 thr / 8 waves. blockIdx.x = head, blockIdx.y = 128-query block.
// Q/K/V tiles staged by the Tensor Data Mover (wave 0 issues, TENSORcnt +
// workgroup barrier synchronize); TDM hardware padding reproduces the
// bank-friendly 16B row pad of the LDS tiles.
__global__ __launch_bounds__(256) void attn_kernel(
    const uint16_t* __restrict__ Q, const uint16_t* __restrict__ K,
    const uint16_t* __restrict__ Vt, float* __restrict__ out) {
    __shared__ uint16_t Qs [128][HD + PAD];  // 34 KB, 272B pitch
    __shared__ uint16_t Ks [32][HD + PAD];   // 8.5 KB, 272B pitch
    __shared__ uint16_t Vst[128][32 + PAD];  // 10 KB, 80B pitch (Vst[d][key])
    __shared__ uint16_t Ps [128][32 + PAD];  // 10 KB
    const int t    = threadIdx.x;
    const int lane = t & 31;
    const int wid  = t >> 5;
    const int ln   = lane & 15;
    const int hi   = lane >> 4;
    const int h    = blockIdx.x;
    const int q0   = blockIdx.y * 128;
    const int r0   = wid * 16;

#if USE_TDM
    const uint32_t ldsQ = (uint32_t)(uintptr_t)&Qs[0][0];
    const uint32_t ldsK = (uint32_t)(uintptr_t)&Ks[0][0];
    const uint32_t ldsV = (uint32_t)(uintptr_t)&Vst[0][0];
    if (wid == 0) {
        // Q tile: 128 rows x 128 elems, row stride DIMC; pad 64DW interval, 4DW
        tdm_load_2d(ldsQ, Q + (size_t)q0 * DIMC + (size_t)h * HD,
                    HD, 128, (uint64_t)DIMC, 5u, 3u);
    }
#else
    {
        const int row = t >> 1;
        const int c0  = (t & 1) * 64;
        const size_t gb = (size_t)(q0 + row) * DIMC + (size_t)h * HD + c0;
#pragma unroll
        for (int i = 0; i < 8; ++i)
            *(uint4*)&Qs[row][c0 + i * 8] = *(const uint4*)&Q[gb + i * 8];
    }
#endif

    float m[8], l[8];
    v8f acc[8];
#pragma unroll
    for (int j = 0; j < 8; ++j) { m[j] = -3.0e38f; l[j] = 0.0f; acc[j] = vzero8(); }

#if !USE_TDM
    const int krow = t >> 3;             // 0..31  (key rows for K staging)
    const int kc0  = (t & 7) * 16;       // 0..112
    const int vrow = t >> 1;             // 0..127 (d rows for V staging)
    const int vc0  = (t & 1) * 16;       // 0 or 16
#endif

    for (int kb = 0; kb < SQ; kb += 32) {
#if USE_TDM
        if (wid == 0) {
            // K tile: 32 keys x 128 d (row stride DIMC); pad 64DW, 4DW
            tdm_load_2d(ldsK, K + (size_t)kb * DIMC + (size_t)h * HD,
                        HD, 32, (uint64_t)DIMC, 5u, 3u);
            // V^T tile: 128 d x 32 keys (row stride SQ); pad 16DW, 4DW
            tdm_load_2d(ldsV, Vt + (size_t)(h * HD) * SQ + kb,
                        32, HD, (uint64_t)SQ, 3u, 3u);
            __builtin_amdgcn_s_wait_tensorcnt(0);   // also covers the Q tile
        }
#else
        {
            const size_t gk = (size_t)(kb + krow) * DIMC + (size_t)h * HD + kc0;
            *(uint4*)&Ks[krow][kc0]     = *(const uint4*)&K[gk];
            *(uint4*)&Ks[krow][kc0 + 8] = *(const uint4*)&K[gk + 8];
            const size_t gv = (size_t)(h * HD + vrow) * SQ + kb + vc0;
            *(uint4*)&Vst[vrow][vc0]     = *(const uint4*)&Vt[gv];
            *(uint4*)&Vst[vrow][vc0 + 8] = *(const uint4*)&Vt[gv + 8];
            if (kb + 32 < SQ) {
                __builtin_prefetch(&K[gk + 32 * DIMC], 0, 0);
                __builtin_prefetch(&Vt[gv + 32], 0, 0);
            }
        }
#endif
        __syncthreads();

        // S = Q (16x128) @ K^T: two 16-key tiles, 4 d-steps of 32
        v8f sc0 = vzero8(), sc1 = vzero8();
#pragma unroll
        for (int kk = 0; kk < HD; kk += 32) {
            Frag A;
            const int am = r0 + ln;
            const int kh = hi * 16;
            A.q[0] = *(const uint4*)&Qs[am][kk + kh];
            A.q[1] = *(const uint4*)&Qs[am][kk + kh + 8];
            const int khB = hi * 16;
            Frag B0, B1;
            B0.q[0] = *(const uint4*)&Ks[ln][kk + khB];
            B0.q[1] = *(const uint4*)&Ks[ln][kk + khB + 8];
            B1.q[0] = *(const uint4*)&Ks[16 + ln][kk + khB];
            B1.q[1] = *(const uint4*)&Ks[16 + ln][kk + khB + 8];
            sc0 = wmma_bf16(A.v, B0.v, sc0);
            sc1 = wmma_bf16(A.v, B1.v, sc1);
        }

        // Online softmax across the 32 keys (row data lives across 16 lanes)
        float fac[8];
#pragma unroll
        for (int j = 0; j < 8; ++j) {
            const float s0 = sc0[j] * SCALE;
            const float s1 = sc1[j] * SCALE;
            float x = fmaxf(s0, s1);
            x = fmaxf(x, __shfl_xor(x, 1));
            x = fmaxf(x, __shfl_xor(x, 2));
            x = fmaxf(x, __shfl_xor(x, 4));
            x = fmaxf(x, __shfl_xor(x, 8));
            const float mn = fmaxf(m[j], x);
            fac[j] = __expf(m[j] - mn);
            const float p0 = __expf(s0 - mn);
            const float p1 = __expf(s1 - mn);
            float rs = p0 + p1;
            rs += __shfl_xor(rs, 1);
            rs += __shfl_xor(rs, 2);
            rs += __shfl_xor(rs, 4);
            rs += __shfl_xor(rs, 8);
            l[j] = l[j] * fac[j] + rs;
            m[j] = mn;
            const int pr = r0 + j + hi * 8;      // wave-private rows of Ps
            Ps[pr][ln]      = f2bf(p0);
            Ps[pr][16 + ln] = f2bf(p1);
        }
#pragma unroll
        for (int dt = 0; dt < 8; ++dt)
#pragma unroll
            for (int j = 0; j < 8; ++j) acc[dt][j] *= fac[j];

        // O += P (16x32) @ V (32 x 128)
        {
            Frag A;
            const int am = r0 + ln;
            const int kh = hi * 16;
            A.q[0] = *(const uint4*)&Ps[am][kh];
            A.q[1] = *(const uint4*)&Ps[am][kh + 8];
            const int khB = hi * 16;
#pragma unroll
            for (int dt = 0; dt < 8; ++dt) {
                Frag B;
                const int n = dt * 16 + ln;      // output d column
                B.q[0] = *(const uint4*)&Vst[n][khB];
                B.q[1] = *(const uint4*)&Vst[n][khB + 8];
                acc[dt] = wmma_bf16(A.v, B.v, acc[dt]);
            }
        }
        __syncthreads();
    }

    // Epilogue: divide by row sum, store fp32 output [S][DIM]
#pragma unroll
    for (int dt = 0; dt < 8; ++dt)
#pragma unroll
        for (int j = 0; j < 8; ++j) {
            const int srow = q0 + r0 + j + hi * 8;
            const int col  = h * HD + dt * 16 + ln;
            out[(size_t)srow * DIMC + col] = acc[dt][j] / l[j];
        }
}

// ---------------------------------------------------------------------------
extern "C" void kernel_launch(void* const* d_in, const int* in_sizes, int n_in,
                              void* d_out, int out_size, void* d_ws, size_t ws_size,
                              hipStream_t stream) {
    (void)in_sizes; (void)n_in; (void)out_size; (void)ws_size;
    const float* hidden = (const float*)d_in[0];
    const float* wq   = (const float*)d_in[1];
    const float* bq   = (const float*)d_in[2];
    const float* wk   = (const float*)d_in[3];
    const float* bk   = (const float*)d_in[4];
    const float* wv   = (const float*)d_in[5];
    const float* bv   = (const float*)d_in[6];
    const float* gq   = (const float*)d_in[7];
    const float* gk   = (const float*)d_in[8];
    const float* cosb = (const float*)d_in[9];
    const float* sinb = (const float*)d_in[10];
    float* outp = (float*)d_out;

    const size_t SD = (size_t)SQ * DIMC;
    const size_t WW = (size_t)DIMC * DIMC;
    uint16_t* Xb  = (uint16_t*)d_ws;
    uint16_t* Wqt = Xb + SD;
    uint16_t* Wkt = Wqt + WW;
    uint16_t* Wvt = Wkt + WW;
    uint16_t* Qb  = Wvt + WW;
    uint16_t* Kb  = Qb + SD;
    uint16_t* Vb  = Kb + SD;
    uint16_t* Vt  = Vb + SD;

    f32_to_bf16_kernel<<<(int)((SD + 255) / 256), 256, 0, stream>>>(hidden, Xb, (int)SD);
    dim3 tg(DIMC / 32, DIMC / 32);
    transpose_f32_to_bf16<<<tg, 256, 0, stream>>>(wq, Wqt, DIMC, DIMC);
    transpose_f32_to_bf16<<<tg, 256, 0, stream>>>(wk, Wkt, DIMC, DIMC);
    transpose_f32_to_bf16<<<tg, 256, 0, stream>>>(wv, Wvt, DIMC, DIMC);

    dim3 gg(SQ / 64, DIMC / 128);
    gemm_bias_bf16<<<gg, 256, 0, stream>>>(Xb, Wqt, bq, Qb);
    gemm_bias_bf16<<<gg, 256, 0, stream>>>(Xb, Wkt, bk, Kb);
    gemm_bias_bf16<<<gg, 256, 0, stream>>>(Xb, Wvt, bv, Vb);

    rmsnorm_rope_kernel<<<SQ * NH, 128, 0, stream>>>(Qb, gq, cosb, sinb);
    rmsnorm_rope_kernel<<<SQ * NH, 128, 0, stream>>>(Kb, gk, cosb, sinb);

    transpose_bf16<<<dim3(DIMC / 32, SQ / 32), 256, 0, stream>>>(Vb, Vt, SQ, DIMC);

    attn_kernel<<<dim3(NH, SQ / 128), 256, 0, stream>>>(Qb, Kb, Vt, outp);
}
